// DPLSTM_40046275068359
// MI455X (gfx1250) — matrix operand order repaired and checked by reading, not compiled, not run on hardware
//
#include <hip/hip_runtime.h>
#include <stdint.h>

// LSTM: T=512, B=128, D=H=1024.  bf16 WMMA (f32 accum) on gfx1250.
#define T_ 512
#define B_ 128
#define D_ 1024
#define H_ 1024

typedef __bf16 v16bf __attribute__((ext_vector_type(16)));
typedef __bf16 v8bf  __attribute__((ext_vector_type(8)));
typedef float  v8f   __attribute__((ext_vector_type(8)));
typedef unsigned short u16;
typedef unsigned short us4 __attribute__((ext_vector_type(4)));

__device__ __forceinline__ u16 f2bf(float f) {
    union { float f; unsigned u; } c; c.f = f;
    unsigned u = c.u;
    return (u16)((u + 0x7FFFu + ((u >> 16) & 1u)) >> 16);  // RNE
}
__device__ __forceinline__ float sigm_(float x) { return 1.f / (1.f + __expf(-x)); }
__device__ __forceinline__ float tanh_(float x) {
    float e = __expf(2.f * x);
    return (e - 1.f) / (e + 1.f);
}

// ---------------------------------------------------------------------------
// Repack w_ih / w_hh [4H x K=1024] f32 -> bf16 in WMMA B-fragment order.
// B matrix 32x16 bf16 (ISA 7.12.2): lane = (N & 15) + 16*k_half, each lane
// holds K = 8*k_half+0..7 and 16+8*k_half+0..7 of column N.  Packed flat index:
//   ((tile*32 + kc)*32 + lane)*16 + e,   tile = global_col/16, kc = K/32.
// ---------------------------------------------------------------------------
__global__ void pack_weights(const float* __restrict__ wih,
                             const float* __restrict__ whh,
                             u16* __restrict__ wih_pk,
                             u16* __restrict__ whh_pk) {
    int gid = blockIdx.x * blockDim.x + threadIdx.x;   // 0 .. 2*262144-1
    const float* src; u16* dst; int G;
    if (gid < 262144) { src = wih; dst = wih_pk; G = gid; }
    else              { src = whh; dst = whh_pk; G = gid - 262144; }
    int tile = G >> 10;          // /(32*32): 0..255
    int kc   = (G >> 5) & 31;
    int lane = G & 31;
    int n = lane & 15, hs = lane >> 4;
    size_t s = (size_t)(tile * 16 + n) * 1024 + (size_t)(kc * 32 + 8 * hs);
    size_t d = (size_t)G * 16;
#pragma unroll
    for (int e = 0; e < 16; ++e) dst[d + e] = f2bf(src[s + e + (e >= 8 ? 8 : 0)]);
}

// Zero h ping-pong + c state, fold the two biases together.
__global__ void init_state(u16* __restrict__ h0, u16* __restrict__ h1,
                           float* __restrict__ c, float* __restrict__ bias,
                           const float* __restrict__ b_ih,
                           const float* __restrict__ b_hh) {
    int i = blockIdx.x * blockDim.x + threadIdx.x;   // 131072 threads
    if (i < B_ * H_) { h0[i] = 0; h1[i] = 0; c[i] = 0.f; }
    if (i < 4 * H_)  bias[i] = b_ih[i] + b_hh[i];
}

// x [T,B,D] f32 -> bf16 (row-major kept: A fragments are row-contiguous).
__global__ void conv_x_bf16(const float4* __restrict__ x,
                            us4* __restrict__ xbf, int n4) {
    int i = blockIdx.x * blockDim.x + threadIdx.x;
    if (i >= n4) return;
    float4 v = x[i];
    us4 o;
    o[0] = f2bf(v.x); o[1] = f2bf(v.y); o[2] = f2bf(v.z); o[3] = f2bf(v.w);
    xbf[i] = o;
}

// A fragment (16-bit A 16x32, ISA 7.12.2): lane m = lane&15 holds two
// contiguous 8-elem chunks at K+8*hs and K+16+8*hs.
__device__ __forceinline__ v16bf ldA(const u16* p, int kc) {
    v8bf a0 = *(const v8bf*)(p + kc * 32);
    v8bf a1 = *(const v8bf*)(p + kc * 32 + 16);
    return __builtin_shufflevector(a0, a1, 0,1,2,3,4,5,6,7,8,9,10,11,12,13,14,15);
}

#define WMMA_BF16(A, Bm, C) \
    __builtin_amdgcn_wmma_f32_16x16x32_bf16(false, (A), false, (Bm), (short)0, (C), false, false)

// ---------------------------------------------------------------------------
// One LSTM timestep.  grid = 64 blocks (one 16-col h-tile each), 256 threads
// = 8 waves (one 16-row batch tile each).  Each wave owns the i/f/g/o tiles
// for its (rows, cols) so the gate nonlinearity + state update fuse here.
// Block stages its 8 weight tiles (4 gates x {w_ih,w_hh}, 256 KB) into LDS.
// Inner loop is software-pipelined: next-iteration A (global/L2) and B (LDS)
// fragments are fetched into distinct registers under the current 8 WMMAs.
// ---------------------------------------------------------------------------
__global__ __launch_bounds__(256, 1)
void lstm_step(const u16* __restrict__ xbf_t,      // [B,D] bf16 at time t
               const u16* __restrict__ wih_pk,     // packed [256 tiles][32 kc][32 lane][16]
               const u16* __restrict__ whh_pk,
               const float* __restrict__ bias,     // [4H] (b_ih+b_hh)
               const u16* __restrict__ h_in,       // [B,H] bf16 (prev step)
               u16* __restrict__ h_out,            // [B,H] bf16 (this step)
               float* __restrict__ cbuf,           // [B,H] f32 in-place
               float* __restrict__ out_t)          // d_out + t*B*H
{
    extern __shared__ __align__(16) u16 smem[];    // 8*32*32*16 u16 = 256 KB
    const int tid = threadIdx.x;
    const int ct  = blockIdx.x;                    // column tile 0..63

    // --- cooperative stage of this ct's weight slice into LDS (ds writes) ---
    {
        const uint4* gih = (const uint4*)wih_pk;
        const uint4* ghh = (const uint4*)whh_pk;
        uint4* s = (uint4*)smem;                   // 16384 x 16B
        for (int i = tid; i < 16384; i += 256) {
            int grp = i >> 1, half = i & 1;        // 32B fragment groups
            int gwi = grp >> 10;                   // gate*2 + wsel, 0..7
            int kc  = (grp >> 5) & 31;
            int ln  = grp & 31;
            int g   = gwi >> 1;
            size_t gsrc = ((((size_t)(g * 64 + ct)) * 32 + kc) * 32 + ln) * 2 + half;
            s[i] = (gwi & 1) ? ghh[gsrc] : gih[gsrc];
        }
    }
    __syncthreads();

    const int rt   = tid >> 5;                     // wave id = batch row tile
    const int lane = tid & 31;
    const int n    = lane & 15;
    const int hs   = lane >> 4;

    const int rowA = rt * 16 + n;
    const u16* pX = xbf_t + (size_t)rowA * D_ + 8 * hs;
    const u16* pH = h_in  + (size_t)rowA * H_ + 8 * hs;
    const u16* sB = smem + (size_t)lane * 16;      // + gwi*16384 + kc*512

    v8f z = 0.f;
    v8f acc0 = z, acc1 = z, acc2 = z, acc3 = z;    // i, f, g, o

    // prologue: fragments for kc = 0
    v16bf ax = ldA(pX, 0), ah = ldA(pH, 0);
    v16bf b0 = *(const v16bf*)(sB + 0 * 16384);
    v16bf b1 = *(const v16bf*)(sB + 1 * 16384);
    v16bf b2 = *(const v16bf*)(sB + 2 * 16384);
    v16bf b3 = *(const v16bf*)(sB + 3 * 16384);
    v16bf b4 = *(const v16bf*)(sB + 4 * 16384);
    v16bf b5 = *(const v16bf*)(sB + 5 * 16384);
    v16bf b6 = *(const v16bf*)(sB + 6 * 16384);
    v16bf b7 = *(const v16bf*)(sB + 7 * 16384);

    for (int kc = 0; kc < 31; ++kc) {
        const int kn = kc + 1;
        // fetch next-iteration fragments into fresh registers (hidden under WMMAs)
        v16bf axn = ldA(pX, kn), ahn = ldA(pH, kn);
        const u16* sN = sB + (size_t)kn * 512;     // 32 lanes * 16 elems per kc
        v16bf n0 = *(const v16bf*)(sN + 0 * 16384);
        v16bf n1 = *(const v16bf*)(sN + 1 * 16384);
        v16bf n2 = *(const v16bf*)(sN + 2 * 16384);
        v16bf n3 = *(const v16bf*)(sN + 3 * 16384);
        v16bf n4 = *(const v16bf*)(sN + 4 * 16384);
        v16bf n5 = *(const v16bf*)(sN + 5 * 16384);
        v16bf n6 = *(const v16bf*)(sN + 6 * 16384);
        v16bf n7 = *(const v16bf*)(sN + 7 * 16384);

        // ax-group then ah-group: same-acc RAW chains spaced 4 apart,
        // A operand held constant across 4 consecutive WMMAs.
        acc0 = WMMA_BF16(ax, b0, acc0);
        acc1 = WMMA_BF16(ax, b2, acc1);
        acc2 = WMMA_BF16(ax, b4, acc2);
        acc3 = WMMA_BF16(ax, b6, acc3);
        acc0 = WMMA_BF16(ah, b1, acc0);
        acc1 = WMMA_BF16(ah, b3, acc1);
        acc2 = WMMA_BF16(ah, b5, acc2);
        acc3 = WMMA_BF16(ah, b7, acc3);

        ax = axn; ah = ahn;
        b0 = n0; b1 = n1; b2 = n2; b3 = n3;
        b4 = n4; b5 = n5; b6 = n6; b7 = n7;
    }
    // epilogue iteration kc = 31
    acc0 = WMMA_BF16(ax, b0, acc0);
    acc1 = WMMA_BF16(ax, b2, acc1);
    acc2 = WMMA_BF16(ax, b4, acc2);
    acc3 = WMMA_BF16(ax, b6, acc3);
    acc0 = WMMA_BF16(ah, b1, acc0);
    acc1 = WMMA_BF16(ah, b3, acc1);
    acc2 = WMMA_BF16(ah, b5, acc2);
    acc3 = WMMA_BF16(ah, b7, acc3);

    // --- fused gate nonlinearity + state update ---
    // D layout (f32 16x16): lane n = lane&15, VGPR r -> row m = r + 8*hs.
    const int col = ct * 16 + n;                   // h index
    const float bi = bias[0 * H_ + col];
    const float bf = bias[1 * H_ + col];
    const float bg = bias[2 * H_ + col];
    const float bo = bias[3 * H_ + col];

#pragma unroll
    for (int r = 0; r < 8; ++r) {
        const int m = rt * 16 + hs * 8 + r;        // batch row
        const float ig = sigm_(acc0[r] + bi);
        const float fg = sigm_(acc1[r] + bf);
        const float gg = tanh_(acc2[r] + bg);
        const float og = sigm_(acc3[r] + bo);
        const size_t idx = (size_t)m * H_ + col;
        const float cp = cbuf[idx];
        const float cn = fg * cp + ig * gg;
        const float hn = og * tanh_(cn);
        cbuf[idx]  = cn;                           // owned exclusively by this lane
        out_t[idx] = hn;                           // hs[t,b,h] f32
        h_out[idx] = f2bf(hn);                     // next step's A operand
    }
}

// ---------------------------------------------------------------------------
extern "C" void kernel_launch(void* const* d_in, const int* in_sizes, int n_in,
                              void* d_out, int out_size, void* d_ws, size_t ws_size,
                              hipStream_t stream) {
    (void)in_sizes; (void)n_in; (void)out_size; (void)ws_size;
    const float* x    = (const float*)d_in[0];
    const float* w_ih = (const float*)d_in[1];
    const float* b_ih = (const float*)d_in[2];
    const float* w_hh = (const float*)d_in[3];
    const float* b_hh = (const float*)d_in[4];
    float* out = (float*)d_out;

    char* ws = (char*)d_ws;
    size_t off = 0;
    auto carve = [&](size_t bytes) -> void* {
        void* p = ws + off;
        off = (off + bytes + 255) & ~(size_t)255;
        return p;
    };
    u16*   wih_pk = (u16*)carve((size_t)4 * H_ * D_ * sizeof(u16));   // 8 MiB
    u16*   whh_pk = (u16*)carve((size_t)4 * H_ * H_ * sizeof(u16));   // 8 MiB
    float* biassm = (float*)carve((size_t)4 * H_ * sizeof(float));
    u16*   xbf    = (u16*)carve((size_t)T_ * B_ * D_ * sizeof(u16));  // 128 MiB
    u16*   hbf0   = (u16*)carve((size_t)B_ * H_ * sizeof(u16));
    u16*   hbf1   = (u16*)carve((size_t)B_ * H_ * sizeof(u16));
    float* cbuf   = (float*)carve((size_t)B_ * H_ * sizeof(float));

    // Prep: weight repack (bf16, fragment order), state init, x -> bf16.
    pack_weights<<<2048, 256, 0, stream>>>(w_ih, w_hh, wih_pk, whh_pk);
    init_state<<<512, 256, 0, stream>>>(hbf0, hbf1, cbuf, biassm, b_ih, b_hh);
    conv_x_bf16<<<(T_ * B_ * D_ / 4 + 255) / 256, 256, 0, stream>>>(
        (const float4*)x, (us4*)xbf, T_ * B_ * D_ / 4);

    // Serial scan: one launch per timestep; h bf16 ping-pong.
    for (int t = 0; t < T_; ++t) {
        const u16* hin = (t & 1) ? hbf1 : hbf0;
        u16*      hout = (t & 1) ? hbf0 : hbf1;
        lstm_step<<<64, 256, 262144, stream>>>(
            xbf + (size_t)t * B_ * D_, wih_pk, whh_pk, biassm,
            hin, hout, cbuf, out + (size_t)t * B_ * H_);
    }

    // h_n = hs[T-1], c_n = c state.
    hipMemcpyAsync(out + (size_t)T_ * B_ * H_,
                   out + (size_t)(T_ - 1) * B_ * H_,
                   (size_t)B_ * H_ * sizeof(float),
                   hipMemcpyDeviceToDevice, stream);
    hipMemcpyAsync(out + (size_t)T_ * B_ * H_ + (size_t)B_ * H_,
                   cbuf,
                   (size_t)B_ * H_ * sizeof(float),
                   hipMemcpyDeviceToDevice, stream);
}
